// FAN_layer_33062658245145
// MI455X (gfx1250) — matrix operand order repaired and checked by reading, not compiled, not run on hardware
//
#include <hip/hip_runtime.h>

// ---------------------------------------------------------------------------
// Types & helpers
// ---------------------------------------------------------------------------
typedef __attribute__((ext_vector_type(16))) __bf16 v16bf;
typedef __attribute__((ext_vector_type(8)))  float  v8f;
typedef __attribute__((ext_vector_type(4)))  unsigned int v4u;
typedef __attribute__((ext_vector_type(4)))  int v4i;
typedef __attribute__((ext_vector_type(8)))  int v8i;

#if defined(__has_builtin)
#if __has_builtin(__builtin_amdgcn_tensor_load_to_lds)
#define HAVE_TDM 1
#endif
#endif
#ifndef HAVE_TDM
#define HAVE_TDM 0
#endif

union Frag {
    unsigned int   w[8];
    v4u            q[2];
    unsigned short u[16];
    v16bf          bf;
};

__device__ __forceinline__ unsigned short f32_to_bf16(float f) {
    unsigned int x = __builtin_bit_cast(unsigned int, f);
    unsigned int r = x + 0x7FFFu + ((x >> 16) & 1u);   // round-to-nearest-even
    return (unsigned short)(r >> 16);
}

// A-fragment (16x32 bf16) K offset for dword pair p (0..7), lane-half hf
__device__ __forceinline__ int kkA(int p, int hf) {
    return (hf << 4) + (p << 1);
}
// B-fragment (32x16 bf16) K offset for dword pair p (0..7), lane-half hf
__device__ __forceinline__ int kkB(int p, int hf) {
    return (hf << 4) + (p << 1);
}

// Async global->LDS 16B copy (per-lane), GVS addressing: saddr base + vgpr off
__device__ __forceinline__ void async_ld16(unsigned lds_off, unsigned goff,
                                           const unsigned short* base) {
    asm volatile("global_load_async_to_lds_b128 %0, %1, %2"
                 :: "v"(lds_off), "v"(goff), "s"(base)
                 : "memory");
}

#if HAVE_TDM
// TDM: load a (128 rows x 32 elem) bf16 tile of Wt[N][K] starting at
// (n0, k0) into LDS at lds_addr, row stride 80B (64B data + 16B pad).
__device__ __forceinline__ void tdm_load_b_tile(const unsigned short* Wt,
                                                int n0, int K, int N, int k0,
                                                unsigned lds_addr) {
    unsigned long long ga =
        (unsigned long long)(size_t)Wt + ((size_t)n0 * K + k0) * 2u;
    v4u g0;
    g0.x = 1u;                                        // count=1, no gather
    g0.y = lds_addr;                                  // lds_addr
    g0.z = (unsigned)ga;                              // global_addr[31:0]
    g0.w = (unsigned)(ga >> 32) | (2u << 30);         // addr[56:32] | type=2
    v8i g1;
    // data_size=2B(code1), pad_enable, pad_interval=16dw(3), pad_amount=4dw(3)
    g1[0] = (int)((1u << 16) | (1u << 20) | (3u << 22) | (3u << 25));
    g1[1] = (int)(((unsigned)K & 0xFFFFu) << 16);     // tensor_dim0 lo16
    g1[2] = (int)(((unsigned)K >> 16) | (((unsigned)N & 0xFFFFu) << 16));
    g1[3] = (int)(((unsigned)N >> 16) | (32u << 16)); // tile_dim0 = 32
    g1[4] = 128;                                      // tile_dim1=128, dim2=0
    g1[5] = K;                                        // dim0_stride lo32
    g1[6] = 0;                                        // stride hi / dim1_stride
    g1[7] = 0;
    v4i z4 = {0, 0, 0, 0};
#if __clang_major__ >= 23
    v8i z8 = {0, 0, 0, 0, 0, 0, 0, 0};
    __builtin_amdgcn_tensor_load_to_lds(g0, g1, z4, z4, z8, 0);
#else
    __builtin_amdgcn_tensor_load_to_lds(g0, g1, z4, z4, 0);
#endif
}
#endif

// ---------------------------------------------------------------------------
// f32 -> bf16 conversion (grid-stride)
// ---------------------------------------------------------------------------
__global__ void cvt_bf16_k(const float* __restrict__ in,
                           unsigned short* __restrict__ out, int n) {
    int i = blockIdx.x * blockDim.x + threadIdx.x;
    int stride = gridDim.x * blockDim.x;
    for (; i < n; i += stride) out[i] = f32_to_bf16(in[i]);
}

// ---------------------------------------------------------------------------
// f32 W[K][N] -> bf16 Wt[N][K] (one-time weight transpose+convert)
// ---------------------------------------------------------------------------
__global__ void wt_bf16_k(const float* __restrict__ W,
                          unsigned short* __restrict__ Wt, int K, int N) {
    int i = blockIdx.x * blockDim.x + threadIdx.x;
    int total = K * N;
    int stride = gridDim.x * blockDim.x;
    for (; i < total; i += stride) {
        int n = i / K, k = i - n * K;
        Wt[i] = f32_to_bf16(W[(size_t)k * N + n]);
    }
}

// ---------------------------------------------------------------------------
// Tiled bf16 WMMA GEMM:  C[M,N] = A[M,K] @ Wt[N,K]^T + bias (+res).
// BLOCK 128x128x32, 256 threads = 8 waves, wave tile 32x64 (2x4 wmma).
// Double-buffered LDS staged by async global->LDS copies (ASYNCcnt);
// weight tile optionally staged by the Tensor Data Mover (TENSORcnt).
// EPI: 0 = bias only, 1 = bias + f32 residual.
// M, N multiples of 128; K multiple of 32 (true for all call sites here).
// ---------------------------------------------------------------------------
template <int EPI>
__global__ __launch_bounds__(256)
void gemm_bf16_k(const unsigned short* __restrict__ A,   // M x K
                 const unsigned short* __restrict__ Wt,  // N x K
                 const float* __restrict__ bias,
                 const float* __restrict__ res,
                 float* __restrict__ outF,
                 unsigned short* __restrict__ outB,
                 int M, int N, int K) {
    // row stride 40 u16 = 80B (multiple of 16B for b128 alignment)
    __shared__ __attribute__((aligned(16))) unsigned short lA[2][128 * 40];
    __shared__ __attribute__((aligned(16))) unsigned short lB[2][128 * 40];

    const int tid  = threadIdx.x;
    const int lane = tid & 31;
    const int wav  = tid >> 5;
    const int lrow = lane & 15;
    const int hf   = lane >> 4;
    const int wm   = wav >> 1;        // 0..3
    const int wn   = wav & 1;         // 0..1
    const int m0   = blockIdx.y * 128;
    const int n0   = blockIdx.x * 128;

    const unsigned lAoffs[2] = {(unsigned)(size_t)&lA[0][0],
                                (unsigned)(size_t)&lA[1][0]};
    const unsigned lBoffs[2] = {(unsigned)(size_t)&lB[0][0],
                                (unsigned)(size_t)&lB[1][0]};

    v8f c[2][4];
    const v8f vzero = {0.f, 0.f, 0.f, 0.f, 0.f, 0.f, 0.f, 0.f};
#pragma unroll
    for (int i = 0; i < 2; ++i)
#pragma unroll
        for (int j = 0; j < 4; ++j) c[i][j] = vzero;

    // per-stage issue of A (and, without TDM, B) tile chunks: 512 x 16B
    auto issue_stage = [&](int k0, int buf) {
#pragma unroll
        for (int j = 0; j < 2; ++j) {
            int cch = tid + j * 256;
            int row = cch >> 2, seg = cch & 3;
            unsigned lofs = lAoffs[buf] + row * 80 + seg * 16;
            unsigned gofs =
                (unsigned)(((size_t)(m0 + row) * K + k0) * 2) + seg * 16;
            async_ld16(lofs, gofs, A);
#if !HAVE_TDM
            unsigned lofs2 = lBoffs[buf] + row * 80 + seg * 16;
            unsigned gofs2 =
                (unsigned)(((size_t)(n0 + row) * K + k0) * 2) + seg * 16;
            async_ld16(lofs2, gofs2, Wt);
#endif
        }
#if HAVE_TDM
        if (wav == 0) tdm_load_b_tile(Wt, n0, K, N, k0, lBoffs[buf]);
#endif
    };

    const int S = K / 32;
    issue_stage(0, 0);

    for (int s = 0; s < S; ++s) {
        const int cur = s & 1;
        if (s + 1 < S) {
            issue_stage((s + 1) * 32, cur ^ 1);
#if HAVE_TDM
            asm volatile("s_wait_asynccnt 0x2" ::: "memory");
            if (wav == 0) __builtin_amdgcn_s_wait_tensorcnt(1);
#else
            asm volatile("s_wait_asynccnt 0x4" ::: "memory");
#endif
        } else {
            asm volatile("s_wait_asynccnt 0x0" ::: "memory");
#if HAVE_TDM
            if (wav == 0) __builtin_amdgcn_s_wait_tensorcnt(0);
#endif
        }
        __syncthreads();

        const unsigned short* Ab = &lA[cur][0];
        const unsigned short* Bb = &lB[cur][0];

        Frag a[2];
#pragma unroll
        for (int mi = 0; mi < 2; ++mi) {
            const v4u* ar =
                (const v4u*)(Ab + (wm * 32 + mi * 16 + lrow) * 40);
            a[mi].q[0] = ar[hf];         // bytes hf*16       -> K [hf*8, +8)
            a[mi].q[1] = ar[2 + hf];     // bytes 32 + hf*16  -> K [16+hf*8, +8)
        }
#pragma unroll
        for (int ni = 0; ni < 4; ++ni) {
            Frag bfr;
            const v4u* br =
                (const v4u*)(Bb + (wn * 64 + ni * 16 + lrow) * 40);
            bfr.q[0] = br[hf];           // bytes hf*16       -> K [hf*8, +8)
            bfr.q[1] = br[2 + hf];       // bytes 32 + hf*16  -> K [16+hf*8, +8)
#pragma unroll
            for (int mi = 0; mi < 2; ++mi)
                c[mi][ni] = __builtin_amdgcn_wmma_f32_16x16x32_bf16(
                    false, a[mi].bf, false, bfr.bf, (short)0, c[mi][ni],
                    false, false);
        }
        __syncthreads();
    }

#pragma unroll
    for (int mi = 0; mi < 2; ++mi) {
#pragma unroll
        for (int ni = 0; ni < 4; ++ni) {
            int gn = n0 + wn * 64 + ni * 16 + lrow;
            float bv = bias[gn];
#pragma unroll
            for (int r = 0; r < 8; ++r) {
                int gm = m0 + wm * 32 + mi * 16 + hf * 8 + r;
                size_t idx = (size_t)gm * N + gn;
                float v = c[mi][ni][r] + bv;
                if (EPI == 1) v += res[idx];
                if (outF) outF[idx] = v;
                if (outB) outB[idx] = f32_to_bf16(v);
            }
        }
    }
}

// ---------------------------------------------------------------------------
// Channel multi-head attention core (per (b,h), 32 query rows per block).
// Q,K,V,O: bf16 [B*L, D] with head h occupying cols [h*64, h*64+64).
// ACT: 1 = softmax, 0 = LeakyReLU(0.5) on scores.
// ---------------------------------------------------------------------------
template <int ACT>
__global__ __launch_bounds__(256)
void cattn_k(const unsigned short* __restrict__ Q,
             const unsigned short* __restrict__ Km,
             const unsigned short* __restrict__ Vm,
             unsigned short* __restrict__ O) {
    const int L = 256, D = 512;
    const float scale = 0.125f;  // 1/sqrt(64)
    const int bb = blockIdx.z, h = blockIdx.y, q0 = blockIdx.x * 32;
    const int tid = threadIdx.x, lane = tid & 31, wav = tid >> 5;
    const int lrow = lane & 15, hf = lane >> 4;

    __shared__ __attribute__((aligned(16))) unsigned short sQ[32][66];
    __shared__ float sS[32][257];
    __shared__ __attribute__((aligned(16))) unsigned short sP[32][258];

    const size_t rowbase = ((size_t)bb * L) * D + (size_t)h * 64;
    const v8f vzero = {0.f, 0.f, 0.f, 0.f, 0.f, 0.f, 0.f, 0.f};

    for (int i = tid; i < 32 * 64; i += 256) {
        int r = i >> 6, c = i & 63;
        sQ[r][c] = Q[rowbase + (size_t)(q0 + r) * D + c];
    }
    __syncthreads();

    // ---- S = scale * Q K^T  (32 x 256) ----
    {
        const int mi = wav >> 2;            // 0..1 (16-row band)
        const int sbase = (wav & 3) * 64;   // 64-col band, 4 n-tiles
        Frag a[2];
#pragma unroll
        for (int ks = 0; ks < 2; ++ks) {
            const unsigned short* arow = &sQ[mi * 16 + lrow][0];
#pragma unroll
            for (int p = 0; p < 8; ++p)
                a[ks].w[p] = *(const unsigned int*)(arow + ks * 32 + kkA(p, hf));
        }
#pragma unroll
        for (int ni = 0; ni < 4; ++ni) {
            int s = sbase + ni * 16 + lrow;
            const unsigned short* kptr = &Km[rowbase + (size_t)s * D];
            v8f acc = vzero;
#pragma unroll
            for (int ks = 0; ks < 2; ++ks) {
                Frag bfr;
#pragma unroll
                for (int p = 0; p < 8; ++p)
                    bfr.w[p] = *(const unsigned int*)(kptr + ks * 32 + kkB(p, hf));
                acc = __builtin_amdgcn_wmma_f32_16x16x32_bf16(
                    false, a[ks].bf, false, bfr.bf, (short)0, acc, false, false);
            }
#pragma unroll
            for (int r = 0; r < 8; ++r)
                sS[mi * 16 + hf * 8 + r][sbase + ni * 16 + lrow] = acc[r] * scale;
        }
    }
    __syncthreads();

    // ---- activation -> bf16 probabilities ----
    if (ACT == 1) {
        for (int r4 = 0; r4 < 4; ++r4) {
            int row = wav * 4 + r4;
            float m = -3.4e38f;
            for (int j = lane; j < 256; j += 32) m = fmaxf(m, sS[row][j]);
#pragma unroll
            for (int off = 16; off; off >>= 1) m = fmaxf(m, __shfl_xor(m, off, 32));
            float sum = 0.f;
            for (int j = lane; j < 256; j += 32) {
                float e = __expf(sS[row][j] - m);
                sS[row][j] = e;
                sum += e;
            }
#pragma unroll
            for (int off = 16; off; off >>= 1) sum += __shfl_xor(sum, off, 32);
            float inv = 1.0f / sum;
            for (int j = lane; j < 256; j += 32)
                sP[row][j] = f32_to_bf16(sS[row][j] * inv);
        }
    } else {
        for (int i = tid; i < 32 * 256; i += 256) {
            int r = i >> 8, c = i & 255;
            float v = sS[r][c];
            sP[r][c] = f32_to_bf16(v >= 0.f ? v : 0.5f * v);
        }
    }
    __syncthreads();

    // ---- O = P @ V  (32 x 64), one 16x16 tile per wave ----
    {
        const int mi = wav >> 2;        // 0..1
        const int ni = wav & 3;         // 0..3
        const int ecol = ni * 16 + lrow;
        v8f acc = vzero;
        for (int ks = 0; ks < 8; ++ks) {
            Frag a, bfr;
            const unsigned short* prow = &sP[mi * 16 + lrow][0];
#pragma unroll
            for (int p = 0; p < 8; ++p)
                a.w[p] = *(const unsigned int*)(prow + ks * 32 + kkA(p, hf));
#pragma unroll
            for (int i = 0; i < 16; ++i) {
                int s = ks * 32 + hf * 16 + i;
                bfr.u[i] = Vm[rowbase + (size_t)s * D + ecol];
            }
            acc = __builtin_amdgcn_wmma_f32_16x16x32_bf16(
                false, a.bf, false, bfr.bf, (short)0, acc, false, false);
        }
#pragma unroll
        for (int r = 0; r < 8; ++r)
            O[rowbase + (size_t)(q0 + mi * 16 + hf * 8 + r) * D + ecol] =
                f32_to_bf16(acc[r]);
    }
}

// ---------------------------------------------------------------------------
// (optional ELU) + LayerNorm, f32 in -> bf16 out. One block per row.
// ---------------------------------------------------------------------------
template <int ELU>
__global__ __launch_bounds__(256)
void layernorm_k(const float* __restrict__ in,
                 const float* __restrict__ g, const float* __restrict__ b,
                 unsigned short* __restrict__ outB, int D) {
    __shared__ float sx[2048];
    __shared__ float sred[256];
    const int row = blockIdx.x, tid = threadIdx.x;
    float ls = 0.f, lss = 0.f;
    for (int j = tid; j < D; j += 256) {
        float x = in[(size_t)row * D + j];
        if (ELU) x = (x > 0.f) ? x : (__expf(x) - 1.f);
        sx[j] = x;
        ls += x;
        lss += x * x;
    }
    sred[tid] = ls;
    __syncthreads();
    for (int s = 128; s > 0; s >>= 1) {
        if (tid < s) sred[tid] += sred[tid + s];
        __syncthreads();
    }
    float mean = sred[0] / (float)D;
    __syncthreads();
    sred[tid] = lss;
    __syncthreads();
    for (int s = 128; s > 0; s >>= 1) {
        if (tid < s) sred[tid] += sred[tid + s];
        __syncthreads();
    }
    float var = sred[0] / (float)D - mean * mean;
    float inv = rsqrtf(var + 1e-5f);
    for (int j = tid; j < D; j += 256)
        outB[(size_t)row * D + j] =
            f32_to_bf16((sx[j] - mean) * inv * g[j] + b[j]);
}

// ---------------------------------------------------------------------------
// Pointwise attention core (HP=16 heads): per row n,
//   A = softmax(qs ks^T / 4),  V = A vs.  Outputs V as bf16 (n, PRED).
// Tiny K (16) -> VALU f32; ~1% of total FLOPs.
// ---------------------------------------------------------------------------
template <int PRED, int SEQ>
__global__ __launch_bounds__(128)
void pwattn_k(const float* __restrict__ q, const float* __restrict__ k,
              const float* __restrict__ v, unsigned short* __restrict__ Vout) {
    const int SH = SEQ / 16;
    const int LH = PRED / 16;
    const int row = blockIdx.x, tid = threadIdx.x;
    __shared__ float sk[SEQ];
    __shared__ float sv[SEQ];
    for (int i = tid; i < SEQ; i += 128) {
        sk[i] = k[(size_t)row * SEQ + i];
        sv[i] = v[(size_t)row * SEQ + i];
    }
    __syncthreads();
    for (int l = tid; l < LH; l += 128) {
        float ql[16];
#pragma unroll
        for (int e = 0; e < 16; ++e) ql[e] = q[(size_t)row * PRED + l * 16 + e];
        float sc[SH];
        float mx = -3.4e38f;
#pragma unroll 4
        for (int s = 0; s < SH; ++s) {
            float d = 0.f;
#pragma unroll
            for (int e = 0; e < 16; ++e) d += ql[e] * sk[s * 16 + e];
            d *= 0.25f;  // 1/sqrt(16)
            sc[s] = d;
            mx = fmaxf(mx, d);
        }
        float sum = 0.f;
#pragma unroll 4
        for (int s = 0; s < SH; ++s) {
            float e = __expf(sc[s] - mx);
            sc[s] = e;
            sum += e;
        }
        float inv = 1.f / sum;
        float acc[16];
#pragma unroll
        for (int e = 0; e < 16; ++e) acc[e] = 0.f;
#pragma unroll 4
        for (int s = 0; s < SH; ++s) {
            float w = sc[s] * inv;
#pragma unroll
            for (int e = 0; e < 16; ++e) acc[e] += w * sv[s * 16 + e];
        }
#pragma unroll
        for (int e = 0; e < 16; ++e)
            Vout[(size_t)row * PRED + l * 16 + e] = f32_to_bf16(acc[e]);
    }
}

// ---------------------------------------------------------------------------
// Host orchestration
// ---------------------------------------------------------------------------
extern "C" void kernel_launch(void* const* d_in, const int* in_sizes, int n_in,
                              void* d_out, int out_size, void* d_ws,
                              size_t ws_size, hipStream_t stream) {
    (void)n_in; (void)out_size; (void)ws_size;
    const int Bb = 112, Ll = 256, Dd = 512, DFF = 2048;
    const int Nrow = Bb * Ll;  // 28672

    char* ws = (char*)d_ws;
    size_t off = 0;
    auto alloc = [&](size_t bytes) -> void* {
        void* p = ws + off;
        off += (bytes + 255) & ~((size_t)255);
        return p;
    };

    const float* x = (const float*)d_in[0];
    // weight input indices (W); bias = index + 1
    const int wIdx[16] = {1, 3, 5, 7,  9, 11, 13, 15,
                          17, 19, 21, 23,  25, 27, 29, 31};
    // per-weight (din=K, dout=N)
    const int wK[16] = {512, 512, 512, 512,  512, 512, 512, 512,
                        512, 512, 512, 2048, 2048, 2048, 2048, 512};
    const int wN[16] = {512, 512, 512, 512,  512, 512, 512, 512,
                        2048, 512, 512, 2048, 512, 2048, 2048, 512};
    unsigned short* wbf[16];
    for (int i = 0; i < 16; ++i)
        wbf[i] = (unsigned short*)alloc((size_t)in_sizes[wIdx[i]] * 2);

    unsigned short* xb   = (unsigned short*)alloc((size_t)Nrow * Dd * 2);
    unsigned short* qb   = (unsigned short*)alloc((size_t)Nrow * Dd * 2);
    unsigned short* kb   = (unsigned short*)alloc((size_t)Nrow * Dd * 2);
    unsigned short* vb   = (unsigned short*)alloc((size_t)Nrow * Dd * 2);
    unsigned short* vat  = (unsigned short*)alloc((size_t)Nrow * Dd * 2);
    unsigned short* x1b  = (unsigned short*)alloc((size_t)Nrow * Dd * 2);
    float*          xres = (float*)alloc((size_t)Nrow * Dd * 4);
    unsigned short* ybf  = (unsigned short*)alloc((size_t)Nrow * Dd * 2);
    float*          qf   = (float*)alloc((size_t)Nrow * DFF * 4);
    float*          kf   = (float*)alloc((size_t)Nrow * DFF * 4);
    float*          vf   = (float*)alloc((size_t)Nrow * DFF * 4);
    unsigned short* vpw  = (unsigned short*)alloc((size_t)Nrow * DFF * 2);
    unsigned short* y2bf = (unsigned short*)alloc((size_t)Nrow * DFF * 2);
    float*          qf2  = (float*)alloc((size_t)Nrow * Dd * 4);
    unsigned short* vpw2 = (unsigned short*)alloc((size_t)Nrow * Dd * 2);

    // --- convert x (row-major) and transpose+convert all weights ---
    {
        int n = Nrow * Dd;
        cvt_bf16_k<<<(n + 255) / 256, 256, 0, stream>>>(x, xb, n);
    }
    for (int i = 0; i < 16; ++i) {
        int n = in_sizes[wIdx[i]];
        wt_bf16_k<<<(n + 255) / 256, 256, 0, stream>>>(
            (const float*)d_in[wIdx[i]], wbf[i], wK[i], wN[i]);
    }

    dim3 blk(256);
    auto gemm0 = [&](const unsigned short* A, int wi, float* oF,
                     unsigned short* oB, int M, int N, int K) {
        dim3 grid(N / 128, M / 128);
        gemm_bf16_k<0><<<grid, blk, 0, stream>>>(
            A, wbf[wi], (const float*)d_in[wIdx[wi] + 1], nullptr, oF, oB, M, N, K);
    };
    auto gemm1 = [&](const unsigned short* A, int wi, const float* res,
                     float* oF, unsigned short* oB, int M, int N, int K) {
        dim3 grid(N / 128, M / 128);
        gemm_bf16_k<1><<<grid, blk, 0, stream>>>(
            A, wbf[wi], (const float*)d_in[wIdx[wi] + 1], res, oF, oB, M, N, K);
    };

    // ---- cattn1 (softmax) ----
    gemm0(xb, 0, nullptr, qb, Nrow, 512, 512);
    gemm0(xb, 1, nullptr, kb, Nrow, 512, 512);
    gemm0(xb, 2, nullptr, vb, Nrow, 512, 512);
    cattn_k<1><<<dim3(Ll / 32, 8, Bb), blk, 0, stream>>>(qb, kb, vb, vat);
    gemm0(vat, 3, nullptr, x1b, Nrow, 512, 512);

    // ---- cattn2 (leakyrelu) + residual with original x ----
    gemm0(x1b, 4, nullptr, qb, Nrow, 512, 512);
    gemm0(x1b, 5, nullptr, kb, Nrow, 512, 512);
    gemm0(x1b, 6, nullptr, vb, Nrow, 512, 512);
    cattn_k<0><<<dim3(Ll / 32, 8, Bb), blk, 0, stream>>>(qb, kb, vb, vat);
    gemm1(vat, 7, x, xres, nullptr, Nrow, 512, 512);

    // ---- norm1 ----
    layernorm_k<0><<<Nrow, 256, 0, stream>>>(
        xres, (const float*)d_in[33], (const float*)d_in[34], ybf, 512);

    // ---- ff1 pwattn (seq=512, pred=2048) ----
    gemm0(ybf, 8,  qf, nullptr, Nrow, 2048, 512);
    gemm0(ybf, 9,  kf, nullptr, Nrow, 512, 512);
    gemm0(ybf, 10, vf, nullptr, Nrow, 512, 512);
    pwattn_k<2048, 512><<<Nrow, 128, 0, stream>>>(qf, kf, vf, vpw);
    gemm1(vpw, 11, qf, qf, nullptr, Nrow, 2048, 2048);  // q + V@Wo+b (in place)

    // ---- elu + norm2 ----
    layernorm_k<1><<<Nrow, 256, 0, stream>>>(
        qf, (const float*)d_in[35], (const float*)d_in[36], y2bf, 2048);

    // ---- ff2 pwattn (seq=2048, pred=512) ----
    gemm0(y2bf, 12, qf2, nullptr, Nrow, 512, 2048);
    gemm0(y2bf, 13, kf,  nullptr, Nrow, 2048, 2048);
    gemm0(y2bf, 14, vf,  nullptr, Nrow, 2048, 2048);
    pwattn_k<512, 2048><<<Nrow, 128, 0, stream>>>(qf2, kf, vf, vpw2);
    gemm1(vpw2, 15, qf2, (float*)d_out, nullptr, Nrow, 512, 512);
    // final reshape (-1, n_vars, D) is a view: memory layout already matches.
}
